// ColbertNegativeCELoss_27350351741580
// MI455X (gfx1250) — matrix operand order repaired and checked by reading, not compiled, not run on hardware
//
#include <hip/hip_runtime.h>
#include <hip/hip_bf16.h>

typedef __bf16 bf16_t;
typedef __attribute__((ext_vector_type(16))) __bf16 v16bf;
typedef __attribute__((ext_vector_type(8)))  float  v8f;

#define B_ 64
#define N_ 32
#define S_ 1024
#define D_ 128
#define INV_TEMP 50.0f  // 1 / 0.02

// f32 -> bf16 round-to-nearest-even, pure bit math.
__device__ __forceinline__ bf16_t f2bf(float f) {
  unsigned u = __builtin_bit_cast(unsigned, f);
  unsigned r = u + 0x7FFFu + ((u >> 16) & 1u);
  unsigned short h = (unsigned short)(r >> 16);
  return __builtin_bit_cast(bf16_t, h);
}

// ---------------------------------------------------------------------------
// Pre-pass 1: pack query embeddings into the WMMA A-operand register layout.
// qpack is an array of v16bf indexed [b][ntile(2)][kc(4)][lane(32)].
// Contiguous k order used here (B matches): lanes 0-15 hold K {0..15}, lanes 16-31 hold
// K {16..31} of each 32-wide K chunk; row M = ntile*16 + (lane&15).
// ---------------------------------------------------------------------------
__global__ void __launch_bounds__(256) pack_q_kernel(
    const float* __restrict__ Q, bf16_t* __restrict__ qpack) {
  const int o = blockIdx.x * 256 + threadIdx.x;  // one bf16 element each
  if (o >= B_ * N_ * D_) return;
  const int j    = o & 15;
  const int lane = (o >> 4) & 31;
  const int kc   = (o >> 9) & 3;
  const int nt   = (o >> 11) & 1;
  const int b    = o >> 12;
  const int half = lane >> 4;
  const int m    = nt * 16 + (lane & 15);
  const int k    = kc * 32 + half * 16 + j;
  qpack[o] = f2bf(Q[((size_t)b * N_ + m) * D_ + k]);
}

// ---------------------------------------------------------------------------
// Pre-pass 2: elementwise f32 -> bf16 (row-major), 2 elements per thread.
// ---------------------------------------------------------------------------
__global__ void __launch_bounds__(256) pack_bf16_kernel(
    const float* __restrict__ src, unsigned* __restrict__ dst, int npairs) {
  const int i = blockIdx.x * 256 + threadIdx.x;
  if (i >= npairs) return;
  const float2 v = ((const float2*)src)[i];
  const unsigned lo = __builtin_bit_cast(unsigned short, f2bf(v.x));
  const unsigned hi = __builtin_bit_cast(unsigned short, f2bf(v.y));
  dst[i] = lo | (hi << 16);
}

// ---------------------------------------------------------------------------
// Main compute: one block per (b,c) pair (in-batch) or per b (neg diagonal).
// 256 threads = 8 wave32; wave w owns a 128-doc s-chunk (8 tiles). Each wave
// keeps BOTH 16x128 query-row A-fragment sets in registers, so every B
// fragment load feeds TWO v_wmma_f32_16x16x32_bf16 (1:1 load:wmma ratio).
// Running max over doc tokens is elementwise on the accumulators (fixed
// VGPR-slot/lane-half == fixed query row in the C/D layout).
// ---------------------------------------------------------------------------
__global__ void __launch_bounds__(256) maxsim_kernel(
    const bf16_t* __restrict__ qpack, const bf16_t* __restrict__ docp,
    const bf16_t* __restrict__ negp, float* __restrict__ scores,
    float* __restrict__ negs) {
  __shared__ float lmax[N_][8];

  const int blk = blockIdx.x;
  const bool isNeg = (blk >= B_ * B_);
  int b, cidx;
  const bf16_t* Dbase;
  if (!isNeg) {
    b = blk / B_;
    cidx = blk % B_;
    Dbase = docp + (size_t)cidx * S_ * D_;
  } else {
    b = blk - B_ * B_;
    cidx = b;
    Dbase = negp + (size_t)b * S_ * D_;
  }

  const int tid    = threadIdx.x;
  const int wave   = tid >> 5;  // s-chunk (0..7), 8 tiles each
  const int lane   = tid & 31;
  const int half   = lane >> 4;
  const int l16    = lane & 15;

  // Both A fragment sets (query rows 0..15 and 16..31), one coalesced 32B
  // load per K-chunk each.
  const v16bf* qpv = (const v16bf*)qpack;
  v16bf a0[4], a1[4];
#pragma unroll
  for (int kc = 0; kc < 4; ++kc) {
    a0[kc] = qpv[(((size_t)b * 2 + 0) * 4 + kc) * 32 + lane];
    a1[kc] = qpv[(((size_t)b * 2 + 1) * 4 + kc) * 32 + lane];
  }

  v8f rm0, rm1;
#pragma unroll
  for (int i = 0; i < 8; ++i) { rm0[i] = -3.0e38f; rm1[i] = -3.0e38f; }

  const int tstart = wave * 8, tend = tstart + 8;
  for (int t = tstart; t < tend; ++t) {
    const int s = t * 16 + l16;  // doc column this lane supplies
    const bf16_t* drow = Dbase + (size_t)s * D_;
    // Unconditional speculative prefetch of the next s-tile row (translation
    // failures silently dropped per ISA).
    __builtin_prefetch(drow + 16 * D_, 0, 3);

    v8f acc0 = {}, acc1 = {};
#pragma unroll
    for (int kc = 0; kc < 4; ++kc) {
      // B fragment: 16 contiguous bf16 per lane (32B aligned) -> 2x b128.
      const v16bf bt = *(const v16bf*)(drow + kc * 32 + half * 16);
      acc0 = __builtin_amdgcn_wmma_f32_16x16x32_bf16(false, a0[kc], false, bt,
                                                     (short)0, acc0, false, false);
      acc1 = __builtin_amdgcn_wmma_f32_16x16x32_bf16(false, a1[kc], false, bt,
                                                     (short)0, acc1, false, false);
    }
#pragma unroll
    for (int i = 0; i < 8; ++i) {
      rm0[i] = fmaxf(rm0[i], acc0[i]);
      rm1[i] = fmaxf(rm1[i], acc1[i]);
    }
  }

  // Max over the 16 doc columns held across each 16-lane group.
#pragma unroll
  for (int off = 8; off >= 1; off >>= 1) {
#pragma unroll
    for (int i = 0; i < 8; ++i) {
      rm0[i] = fmaxf(rm0[i], __shfl_xor(rm0[i], off));
      rm1[i] = fmaxf(rm1[i], __shfl_xor(rm1[i], off));
    }
  }
  if (l16 == 0) {
    // half=0 holds rows {0..7 | 16..23}, half=1 holds rows {8..15 | 24..31}.
#pragma unroll
    for (int i = 0; i < 8; ++i) {
      lmax[half * 8 + i][wave]      = rm0[i];
      lmax[16 + half * 8 + i][wave] = rm1[i];
    }
  }
  __syncthreads();

  // Combine the 8 s-chunks per query row, then sum over the 32 rows.
  if (tid < N_) {
    float v = lmax[tid][0];
#pragma unroll
    for (int c = 1; c < 8; ++c) v = fmaxf(v, lmax[tid][c]);
#pragma unroll
    for (int off = 16; off >= 1; off >>= 1) v += __shfl_xor(v, off);
    if (tid == 0) {
      if (!isNeg) scores[b * B_ + cidx] = v;
      else        negs[b] = v;
    }
  }
}

// Tiny epilogue: softplus((neg-pos)/T).mean + CE(log_softmax(scores/T)), /2.
__global__ void __launch_bounds__(64) loss_kernel(
    const float* __restrict__ scores, const float* __restrict__ negs,
    const int* __restrict__ offp, float* __restrict__ out) {
  __shared__ float ssp[B_];
  __shared__ float sce[B_];
  const int b = threadIdx.x;
  const float* row = scores + b * B_;

  const float pos = row[b];
  const float neg = negs[b];
  const float x = (neg - pos) * INV_TEMP;
  const float sp = (x > 0.f) ? x + log1pf(expf(-x)) : log1pf(expf(x));

  float mx = -3.0e38f;
  for (int c = 0; c < B_; ++c) mx = fmaxf(mx, row[c]);
  float sum = 0.f;
  for (int c = 0; c < B_; ++c) sum += expf((row[c] - mx) * INV_TEMP);
  const float lse = mx * INV_TEMP + logf(sum);

  int label = b + offp[0];
  float lp = 0.f;
  if (label >= 0 && label < B_) lp = row[label] * INV_TEMP - lse;

  ssp[b] = sp;
  sce[b] = -lp;
  __syncthreads();
#pragma unroll
  for (int st = 32; st > 0; st >>= 1) {
    if (b < st) {
      ssp[b] += ssp[b + st];
      sce[b] += sce[b + st];
    }
    __syncthreads();
  }
  if (b == 0) out[0] = (ssp[0] / (float)B_ + sce[0] / (float)B_) * 0.5f;
}

extern "C" void kernel_launch(void* const* d_in, const int* in_sizes, int n_in,
                              void* d_out, int out_size, void* d_ws, size_t ws_size,
                              hipStream_t stream) {
  (void)in_sizes; (void)n_in; (void)out_size; (void)ws_size;
  const float* q   = (const float*)d_in[0];
  const float* doc = (const float*)d_in[1];
  const float* ndc = (const float*)d_in[2];
  const int*   off = (const int*)d_in[3];

  // Workspace layout (all sub-buffers 32-byte aligned).
  float*  scores  = (float*)d_ws;                    // 64*64 f32
  float*  negsws  = scores + B_ * B_;                // 64 f32
  bf16_t* qpack   = (bf16_t*)(negsws + B_);          // 256K bf16 (A layout)
  bf16_t* docpack = qpack + (size_t)B_ * N_ * D_;    // 8M bf16 row-major
  bf16_t* negpack = docpack + (size_t)B_ * S_ * D_;  // 8M bf16 row-major

  // Pre-pass: one-time f32 -> bf16 conversion (bandwidth-bound).
  {
    const int nq = B_ * N_ * D_;
    pack_q_kernel<<<(nq + 255) / 256, 256, 0, stream>>>(q, qpack);
    const int npairs = (B_ * S_ * D_) / 2;
    pack_bf16_kernel<<<(npairs + 255) / 256, 256, 0, stream>>>(
        doc, (unsigned*)docpack, npairs);
    pack_bf16_kernel<<<(npairs + 255) / 256, 256, 0, stream>>>(
        ndc, (unsigned*)negpack, npairs);
  }

  maxsim_kernel<<<B_ * B_ + B_, 256, 0, stream>>>(qpack, docpack, negpack,
                                                  scores, negsws);
  loss_kernel<<<1, 64, 0, stream>>>(scores, negsws, off, (float*)d_out);
}
